// StructuralHoleHAN_50182397886683
// MI455X (gfx1250) — matrix operand order repaired and checked by reading, not compiled, not run on hardware
//
#include <hip/hip_runtime.h>
#include <hip/hip_bf16.h>
#include <cstddef>

// ---------------- constants ----------------
#define HHEADS 4
#define DDIM   32
#define HIDC   128
#define OUTC   64

typedef __attribute__((ext_vector_type(16))) __bf16 v16bf;
typedef __attribute__((ext_vector_type(8)))  float  v8f;
typedef int vint4 __attribute__((__vector_size__(16)));

union Frag16 {
  v16bf  v;
  __bf16 e[16];
};

// ---------------- CDNA5 async global->LDS staging (guarded) ----------------
#if defined(__has_builtin)
#  if __has_builtin(__builtin_amdgcn_global_load_async_to_lds_b128)
#    define HAVE_ASYNC_LDS 1
#  else
#    define HAVE_ASYNC_LDS 0
#  endif
#  if __has_builtin(__builtin_amdgcn_s_wait_asynccnt)
#    define WAIT_ASYNC(n) __builtin_amdgcn_s_wait_asynccnt(n)
#  else
#    define WAIT_ASYNC(n) asm volatile("s_wait_asynccnt %0" ::"n"(n) : "memory")
#  endif
#else
#  define HAVE_ASYNC_LDS 0
#  define WAIT_ASYNC(n)
#endif

__device__ __forceinline__ void async_cp16(const float* g, float* l) {
#if HAVE_ASYNC_LDS
  __builtin_amdgcn_global_load_async_to_lds_b128(
      (__attribute__((address_space(1))) vint4*)g,
      (__attribute__((address_space(3))) vint4*)l, 0, 0);
#else
  (void)g; (void)l;
#endif
}

// ---------------- small utility kernels ----------------
__global__ void fill_k(float* __restrict__ p, long n, float v) {
  long i = (long)blockIdx.x * blockDim.x + threadIdx.x;
  if (i < n) p[i] = v;
}

__global__ void relu_k(float* __restrict__ p, long n) {
  long i = (long)blockIdx.x * blockDim.x + threadIdx.x;
  if (i < n) p[i] = fmaxf(p[i], 0.0f);
}

__global__ void elu_k(const float* __restrict__ in, float* __restrict__ out, long n) {
  long i = (long)blockIdx.x * blockDim.x + threadIdx.x;
  if (i < n) { float x = in[i]; out[i] = x > 0.0f ? x : expm1f(x); }
}

__global__ void combine2_elu_k(const float* __restrict__ o0, const float* __restrict__ o1,
                               const float* __restrict__ w, float* __restrict__ out, long n) {
  long i = (long)blockIdx.x * blockDim.x + threadIdx.x;
  if (i < n) {
    float x = w[0] * o0[i] + w[1] * o1[i];
    out[i] = x > 0.0f ? x : expm1f(x);
  }
}

__device__ __forceinline__ void atomicMaxF(float* addr, float val) {
  unsigned* ua = reinterpret_cast<unsigned*>(addr);
  unsigned cur = *(volatile unsigned*)ua;
  while (__uint_as_float(cur) < val) {
    unsigned assumed = cur;
    cur = atomicCAS(ua, assumed, __float_as_uint(val));
    if (cur == assumed) break;
  }
}

// =====================================================================
// WMMA bf16 GEMM core:  C = A(NxK) @ W(KxM) + bias     (N % 16 == 0)
// MT waves per block, wave w owns 16-col slice; M = MT*16.
// A tile = full 16 x K f32 slab (contiguous in A), double-buffered in LDS,
// staged with async global->LDS b128 when available. B fragments for all
// K-steps hoisted into registers once per block (grid-stride over tiles).
// =====================================================================
template <int K, int MT, bool TANH>
__global__ __launch_bounds__(32 * MT)
void gemm_core(const float* __restrict__ A, const float* __restrict__ W,
               const float* __restrict__ bias, float* __restrict__ Cout,
               int ntiles) {
  constexpr int M     = MT * 16;
  constexpr int KS    = K / 32;            // WMMA K-steps
  constexpr int TILEF = 16 * K;            // floats per A tile
  constexpr int NASY  = TILEF / 128;       // b128 ops per tile (4 floats/lane * 32 lanes)
  constexpr int SPW   = NASY / MT;         // async issues per wave per tile
  constexpr int NT    = 32 * MT;           // threads
  constexpr int F4PT  = TILEF / 4 / NT;    // float4 per thread (sync fallback)

  __shared__ float Atile[2][TILEF];

  const int lane = threadIdx.x & 31;
  const int wave = threadIdx.x >> 5;
  const int mrow = lane & 15;
  const int grp  = lane >> 4;
  const int col  = wave * 16 + mrow;

  // ---- hoist B fragments (all K-steps) into registers ----
  Frag16 bfr[KS];
#pragma unroll
  for (int ks = 0; ks < KS; ++ks) {
    const int kb = ks * 32 + grp * 16;
#pragma unroll
    for (int i = 0; i < 16; ++i)
      bfr[ks].e[i] = (__bf16)W[(size_t)(kb - grp * 8 + (i & 7) + (i >> 3) * 16) * M + col];
  }

  const int G = gridDim.x;

  // ---- staging helper ----
  auto stage = [&](int b, int t) {
    const float* gp = A + (size_t)t * TILEF;  // 16 full rows are contiguous
#if HAVE_ASYNC_LDS
#pragma unroll
    for (int j = 0; j < SPW; ++j) {
      int f = ((wave * SPW + j) * 32 + lane) * 4;
      async_cp16(gp + f, &Atile[b][f]);
    }
#else
#pragma unroll
    for (int j = 0; j < F4PT; ++j) {
      int f = (j * NT + threadIdx.x) * 4;
      *(float4*)&Atile[b][f] = *(const float4*)(gp + f);
    }
#endif
  };

  int t = blockIdx.x;
  if (t >= ntiles) return;
  int buf = 0;
  stage(0, t);

  for (; t < ntiles; t += G) {
    const int nt = t + G;
    if (nt < ntiles) {
      stage(buf ^ 1, nt);
      WAIT_ASYNC(SPW);     // current tile's async copies done (per wave)
    } else {
      WAIT_ASYNC(0);
    }
    __syncthreads();

    v8f acc = {};
    const float* arow = &Atile[buf][mrow * K];
#pragma unroll
    for (int ks = 0; ks < KS; ++ks) {
      const float* p0 = arow + ks * 32 + grp * 8;         // K = g*8 .. g*8+7
      const float* p1 = arow + ks * 32 + 16 + grp * 8;    // K = 16+g*8 .. +7
      float4 f0 = *(const float4*)p0;
      float4 f1 = *(const float4*)(p0 + 4);
      float4 f2 = *(const float4*)p1;
      float4 f3 = *(const float4*)(p1 + 4);
      Frag16 a;
      a.e[0]  = (__bf16)f0.x; a.e[1]  = (__bf16)f0.y;
      a.e[2]  = (__bf16)f0.z; a.e[3]  = (__bf16)f0.w;
      a.e[4]  = (__bf16)f1.x; a.e[5]  = (__bf16)f1.y;
      a.e[6]  = (__bf16)f1.z; a.e[7]  = (__bf16)f1.w;
      a.e[8]  = (__bf16)f2.x; a.e[9]  = (__bf16)f2.y;
      a.e[10] = (__bf16)f2.z; a.e[11] = (__bf16)f2.w;
      a.e[12] = (__bf16)f3.x; a.e[13] = (__bf16)f3.y;
      a.e[14] = (__bf16)f3.z; a.e[15] = (__bf16)f3.w;
      acc = __builtin_amdgcn_wmma_f32_16x16x32_bf16(false, a.v, false, bfr[ks].v,
                                                    (short)0, acc, false, false);
    }

    const float bv = bias[col];
    if (TANH) {
      float sum = 0.0f;
#pragma unroll
      for (int j = 0; j < 8; ++j) sum += tanhf(acc[j] + bv);
      atomicAdd(&Cout[col], sum);
    } else {
#pragma unroll
      for (int j = 0; j < 8; ++j) {
        int row = t * 16 + j + 8 * grp;
        Cout[(size_t)row * M + col] = acc[j] + bv;
      }
    }
    __syncthreads();
    buf ^= 1;
  }
}

// ---------------- GAT edge attention ----------------
__global__ void node_score_k(const float* __restrict__ h, const float* __restrict__ a,
                             float* __restrict__ sc, int n) {
  int t = blockIdx.x * blockDim.x + threadIdx.x;
  if (t >= n * HHEADS) return;
  int node = t >> 2, hh = t & 3;
  const float* hp = h + (size_t)node * HIDC + hh * DDIM;
  const float* ap = a + hh * DDIM;
  float s = 0.0f;
#pragma unroll
  for (int d = 0; d < DDIM; ++d) s += hp[d] * ap[d];
  sc[t] = s;
}

__global__ void edge_logit_max_k(const int* __restrict__ src, const int* __restrict__ dst,
                                 const float* __restrict__ ss, const float* __restrict__ sd,
                                 float* __restrict__ ebuf, float* __restrict__ m, int E) {
  int e = blockIdx.x * blockDim.x + threadIdx.x;
  if (e >= E) return;
  int s_ = src[e], d_ = dst[e];
#pragma unroll
  for (int h = 0; h < HHEADS; ++h) {
    float l = ss[s_ * HHEADS + h] + sd[d_ * HHEADS + h];
    l = l > 0.0f ? l : 0.2f * l;
    ebuf[(size_t)e * HHEADS + h] = l;
    atomicMaxF(&m[d_ * HHEADS + h], l);
  }
}

__global__ void edge_exp_sum_k(const int* __restrict__ dst, float* __restrict__ ebuf,
                               const float* __restrict__ m, float* __restrict__ ssum, int E) {
  int e = blockIdx.x * blockDim.x + threadIdx.x;
  if (e >= E) return;
  int d_ = dst[e];
#pragma unroll
  for (int h = 0; h < HHEADS; ++h) {
    float l = ebuf[(size_t)e * HHEADS + h];
    float ex = __expf(l - m[d_ * HHEADS + h]);
    ebuf[(size_t)e * HHEADS + h] = ex;
    atomicAdd(&ssum[d_ * HHEADS + h], ex);
  }
}

__global__ void edge_scatter_k(const int* __restrict__ src, const int* __restrict__ dst,
                               const float* __restrict__ hsrc, const float* __restrict__ ebuf,
                               const float* __restrict__ ssum, float* __restrict__ out, int E) {
  int gw = (blockIdx.x * blockDim.x + threadIdx.x) >> 5;
  int lane = threadIdx.x & 31;
  if (gw >= E) return;
  int s_ = src[gw], d_ = dst[gw];
  const float* hp = hsrc + (size_t)s_ * HIDC;
  float* op = out + (size_t)d_ * HIDC;
#pragma unroll
  for (int j = 0; j < HHEADS; ++j) {
    float alpha = ebuf[(size_t)gw * HHEADS + j] / (ssum[d_ * HHEADS + j] + 1e-16f);
    atomicAdd(&op[j * DDIM + lane], hp[j * DDIM + lane] * alpha);
  }
}

// ---------------- semantic attention score + softmax (T=2) ----------------
__global__ void score2_softmax_k(const float* __restrict__ q, const float* __restrict__ colsum,
                                 float* __restrict__ w, int N) {
  if (threadIdx.x != 0 || blockIdx.x != 0) return;
  float s0 = 0.0f, s1 = 0.0f;
  for (int c = 0; c < HIDC; ++c) {
    s0 += q[c] * colsum[c];
    s1 += q[c] * colsum[HIDC + c];
  }
  float inv = 1.0f / (float)N;
  s0 *= inv; s1 *= inv;
  float mx = fmaxf(s0, s1);
  float e0 = __expf(s0 - mx), e1 = __expf(s1 - mx);
  float den = e0 + e1;
  w[0] = e0 / den; w[1] = e1 / den;
}

// ---------------- L2 row normalize (64 cols, one wave per row) ----------------
__global__ void l2norm_rows_k(float* __restrict__ p, int n) {
  int row = (blockIdx.x * blockDim.x + threadIdx.x) >> 5;
  int lane = threadIdx.x & 31;
  if (row >= n) return;
  float* rp = p + (size_t)row * OUTC;
  float v0 = rp[lane], v1 = rp[lane + 32];
  float ss = v0 * v0 + v1 * v1;
#pragma unroll
  for (int off = 16; off > 0; off >>= 1) ss += __shfl_xor(ss, off, 32);
  float nrm = fmaxf(sqrtf(ss), 1e-12f);
  rp[lane] = v0 / nrm;
  rp[lane + 32] = v1 / nrm;
}

// ---------------- host orchestration ----------------
static inline int cdiv(long a, long b) { return (int)((a + b - 1) / b); }
static inline int imin(int a, int b) { return a < b ? a : b; }

static void launch_linear(const float* A, const float* W, const float* b, float* C,
                          int N, int K, int M, hipStream_t s) {
  int ntiles = N / 16;                    // N is a multiple of 16 here
  int grid = imin(ntiles, 2048);
  if (K == 256 && M == 128)
    gemm_core<256, 8, false><<<grid, 256, 0, s>>>(A, W, b, C, ntiles);
  else if (K == 128 && M == 128)
    gemm_core<128, 8, false><<<grid, 256, 0, s>>>(A, W, b, C, ntiles);
  else  // K == 128, M == 64 (projection)
    gemm_core<128, 4, false><<<grid, 128, 0, s>>>(A, W, b, C, ntiles);
}

static void launch_tanh_colsum(const float* A, const float* W, const float* b, float* colsum,
                               int N, hipStream_t s) {
  int ntiles = N / 16;
  int grid = imin(ntiles, 2048);
  gemm_core<128, 8, true><<<grid, 256, 0, s>>>(A, W, b, colsum, ntiles);
}

static void run_edge_att(const int* ei, int E,
                         const float* hsrc, const float* hdst,
                         const float* a_s, const float* a_d,
                         int n_src, int n_dst,
                         float* ss, float* sd, float* m, float* ssum, float* ebuf,
                         float* out, hipStream_t s) {
  const int* src = ei;
  const int* dst = ei + E;
  node_score_k<<<cdiv((long)n_src * HHEADS, 256), 256, 0, s>>>(hsrc, a_s, ss, n_src);
  node_score_k<<<cdiv((long)n_dst * HHEADS, 256), 256, 0, s>>>(hdst, a_d, sd, n_dst);
  fill_k<<<cdiv((long)n_dst * HHEADS, 256), 256, 0, s>>>(m, (long)n_dst * HHEADS, -1e30f);
  fill_k<<<cdiv((long)n_dst * HHEADS, 256), 256, 0, s>>>(ssum, (long)n_dst * HHEADS, 0.0f);
  fill_k<<<cdiv((long)n_dst * HIDC, 256), 256, 0, s>>>(out, (long)n_dst * HIDC, 0.0f);
  edge_logit_max_k<<<cdiv(E, 256), 256, 0, s>>>(src, dst, ss, sd, ebuf, m, E);
  edge_exp_sum_k<<<cdiv(E, 256), 256, 0, s>>>(dst, ebuf, m, ssum, E);
  edge_scatter_k<<<cdiv(E, 8), 256, 0, s>>>(src, dst, hsrc, ebuf, ssum, out, E);
  relu_k<<<cdiv((long)n_dst * HIDC, 256), 256, 0, s>>>(out, (long)n_dst * HIDC);
}

extern "C" void kernel_launch(void* const* d_in, const int* in_sizes, int n_in,
                              void* d_out, int out_size, void* d_ws, size_t ws_size,
                              hipStream_t stream) {
  const float* xc = (const float*)d_in[0];
  const float* xd = (const float*)d_in[1];
  const int* ei_cc = (const int*)d_in[2];
  const int* ei_dc = (const int*)d_in[3];
  const int* ei_cd = (const int*)d_in[4];
  const float* W1c = (const float*)d_in[5];  const float* b1c = (const float*)d_in[6];
  const float* W1d = (const float*)d_in[7];  const float* b1d = (const float*)d_in[8];
  const float* a1s_cc = (const float*)d_in[9];  const float* a1d_cc = (const float*)d_in[10];
  const float* a1s_dc = (const float*)d_in[11]; const float* a1d_dc = (const float*)d_in[12];
  const float* a1s_cd = (const float*)d_in[13]; const float* a1d_cd = (const float*)d_in[14];
  const float* k1W = (const float*)d_in[15]; const float* k1b = (const float*)d_in[16];
  const float* q1  = (const float*)d_in[17];
  const float* W2c = (const float*)d_in[18]; const float* b2c = (const float*)d_in[19];
  const float* W2d = (const float*)d_in[20]; const float* b2d = (const float*)d_in[21];
  const float* a2s_cc = (const float*)d_in[22]; const float* a2d_cc = (const float*)d_in[23];
  const float* a2s_dc = (const float*)d_in[24]; const float* a2d_dc = (const float*)d_in[25];
  const float* a2s_cd = (const float*)d_in[26]; const float* a2d_cd = (const float*)d_in[27];
  const float* k2W = (const float*)d_in[28]; const float* k2b = (const float*)d_in[29];
  const float* q2  = (const float*)d_in[30];
  const float* pW  = (const float*)d_in[31]; const float* pb = (const float*)d_in[32];

  const int FC = 256;
  const int NC = in_sizes[0] / FC;
  const int ND = in_sizes[1] / FC;
  const int Ecc = in_sizes[2] / 2;
  const int Edc = in_sizes[3] / 2;
  const int Ecd = in_sizes[4] / 2;
  const int maxN = NC > ND ? NC : ND;
  int maxE = Ecc; if (Edc > maxE) maxE = Edc; if (Ecd > maxE) maxE = Ecd;

  float* base = (float*)d_ws;
  size_t off = 0;
  auto alloc = [&](size_t n) { float* p = base + off; off += n; return p; };
  float* hc     = alloc((size_t)NC * HIDC);
  float* hd     = alloc((size_t)ND * HIDC);
  float* o_cc   = alloc((size_t)NC * HIDC);
  float* o_dc   = alloc((size_t)NC * HIDC);
  float* o_cd   = alloc((size_t)ND * HIDC);
  float* c1     = alloc((size_t)NC * HIDC);
  float* d1     = alloc((size_t)ND * HIDC);
  float* ss     = alloc((size_t)maxN * HHEADS);
  float* sd     = alloc((size_t)maxN * HHEADS);
  float* m      = alloc((size_t)maxN * HHEADS);
  float* ssum   = alloc((size_t)maxN * HHEADS);
  float* ebuf   = alloc((size_t)maxE * HHEADS);
  float* colsum = alloc(2 * HIDC);
  float* wsem   = alloc(2);
  (void)ws_size;

  // ================= Layer 1 =================
  launch_linear(xc, W1c, b1c, hc, NC, FC, HIDC, stream);
  launch_linear(xd, W1d, b1d, hd, ND, FC, HIDC, stream);

  run_edge_att(ei_cc, Ecc, hc, hc, a1s_cc, a1d_cc, NC, NC, ss, sd, m, ssum, ebuf, o_cc, stream);
  run_edge_att(ei_dc, Edc, hd, hc, a1s_dc, a1d_dc, ND, NC, ss, sd, m, ssum, ebuf, o_dc, stream);
  run_edge_att(ei_cd, Ecd, hc, hd, a1s_cd, a1d_cd, NC, ND, ss, sd, m, ssum, ebuf, o_cd, stream);

  fill_k<<<1, 256, 0, stream>>>(colsum, 2 * HIDC, 0.0f);
  launch_tanh_colsum(o_cc, k1W, k1b, colsum, NC, stream);
  launch_tanh_colsum(o_dc, k1W, k1b, colsum + HIDC, NC, stream);
  score2_softmax_k<<<1, 32, 0, stream>>>(q1, colsum, wsem, NC);
  combine2_elu_k<<<cdiv((long)NC * HIDC, 256), 256, 0, stream>>>(o_cc, o_dc, wsem, c1, (long)NC * HIDC);
  elu_k<<<cdiv((long)ND * HIDC, 256), 256, 0, stream>>>(o_cd, d1, (long)ND * HIDC);

  // ================= Layer 2 =================
  launch_linear(c1, W2c, b2c, hc, NC, HIDC, HIDC, stream);
  launch_linear(d1, W2d, b2d, hd, ND, HIDC, HIDC, stream);

  run_edge_att(ei_cc, Ecc, hc, hc, a2s_cc, a2d_cc, NC, NC, ss, sd, m, ssum, ebuf, o_cc, stream);
  run_edge_att(ei_dc, Edc, hd, hc, a2s_dc, a2d_dc, ND, NC, ss, sd, m, ssum, ebuf, o_dc, stream);
  run_edge_att(ei_cd, Ecd, hc, hd, a2s_cd, a2d_cd, NC, ND, ss, sd, m, ssum, ebuf, o_cd, stream);

  fill_k<<<1, 256, 0, stream>>>(colsum, 2 * HIDC, 0.0f);
  launch_tanh_colsum(o_cc, k2W, k2b, colsum, NC, stream);
  launch_tanh_colsum(o_dc, k2W, k2b, colsum + HIDC, NC, stream);
  score2_softmax_k<<<1, 32, 0, stream>>>(q2, colsum, wsem, NC);
  combine2_elu_k<<<cdiv((long)NC * HIDC, 256), 256, 0, stream>>>(o_cc, o_dc, wsem, c1, (long)NC * HIDC);
  elu_k<<<cdiv((long)ND * HIDC, 256), 256, 0, stream>>>(o_cd, d1, (long)ND * HIDC);

  // ================= Projection + normalize =================
  float* pc = (float*)d_out;
  float* pd = (float*)d_out + (size_t)NC * OUTC;
  launch_linear(c1, pW, pb, pc, NC, HIDC, OUTC, stream);
  launch_linear(d1, pW, pb, pd, ND, HIDC, OUTC, stream);
  l2norm_rows_k<<<cdiv(NC, 8), 256, 0, stream>>>(pc, NC);
}
